// Attention_26456998543544
// MI455X (gfx1250) — matrix operand order repaired and checked by reading, not compiled, not run on hardware
//
#include <hip/hip_runtime.h>
#include <hip/hip_bf16.h>

// ---------------------------------------------------------------------------
// Attention forward for MI455X (gfx1250), bf16 WMMA path.
//   B=2, L=2048, D=2048, H=16, hd=128
// ---------------------------------------------------------------------------

#define B_   2
#define L_   2048
#define D_   2048
#define H_   16
#define HD_  128
#define M_   (B_ * L_)        // 4096 rows for all (M x 2048) GEMMs

typedef __attribute__((ext_vector_type(16))) __bf16 v16bf;
typedef __attribute__((ext_vector_type(8)))  float  v8f;

__device__ __forceinline__ v8f wmma_bf16(v16bf a, v16bf b, v8f c) {
  // (neg_a, A, neg_b, B, c_mod, C, reuse_a, reuse_b)
  return __builtin_amdgcn_wmma_f32_16x16x32_bf16(false, a, false, b,
                                                 (short)0, c, false, false);
}

__device__ __forceinline__ __bf16 f2bf(float f) {
  union { unsigned u; float fl; } in; in.fl = f;
  unsigned u = in.u;
  unsigned r = (u + 0x7FFFu + ((u >> 16) & 1u)) >> 16;  // RNE
  union { unsigned short s; __bf16 b; } out; out.s = (unsigned short)r;
  return out.b;
}

// Load a 16-element bf16 fragment from two 16-byte chunks.
__device__ __forceinline__ v16bf load_frag(const __bf16* p0, const __bf16* p1) {
  union { v16bf v; uint4 u[2]; } r;
  r.u[0] = *(const uint4*)p0;
  r.u[1] = *(const uint4*)p1;
  return r.v;
}

// ---------------------------------------------------------------------------
// CDNA5 async global->LDS staging (ASYNCcnt path), with sync fallback.
// ---------------------------------------------------------------------------
#if __has_builtin(__builtin_amdgcn_global_load_async_to_lds_b128)
#define HAVE_ASYNC_LDS 1
#else
#define HAVE_ASYNC_LDS 0
#endif

typedef int v4i_ __attribute__((vector_size(16)));
typedef __attribute__((address_space(1))) v4i_ gv4i;   // global int4
typedef __attribute__((address_space(3))) v4i_ lv4i;   // LDS int4

__device__ __forceinline__ void wait_async0() {
#if HAVE_ASYNC_LDS
#if __has_builtin(__builtin_amdgcn_s_wait_asynccnt)
  __builtin_amdgcn_s_wait_asynccnt(0);
#else
  asm volatile("s_wait_asynccnt 0x0" ::: "memory");
#endif
#endif
}

// Copy 32 contiguous bytes global -> LDS (per thread).
__device__ __forceinline__ void stage32(const __bf16* __restrict__ g,
                                        __bf16* __restrict__ l) {
#if HAVE_ASYNC_LDS
  gv4i* gp = (gv4i*)g;
  lv4i* lp = (lv4i*)l;
  // IOFFSET applies to BOTH the global and LDS address (ISA 10.x async ops),
  // and both chunks are contiguous in both spaces.
  __builtin_amdgcn_global_load_async_to_lds_b128(gp, lp, 0, 0);
  __builtin_amdgcn_global_load_async_to_lds_b128(gp, lp, 16, 0);
#else
  uint4 a0 = ((const uint4*)g)[0];
  uint4 a1 = ((const uint4*)g)[1];
  ((uint4*)l)[0] = a0;
  ((uint4*)l)[1] = a1;
#endif
}

// ---------------------------------------------------------------------------
// Elementwise f32 -> bf16 cast
// ---------------------------------------------------------------------------
__global__ __launch_bounds__(256) void cast_kernel(const float* __restrict__ src,
                                                   __bf16* __restrict__ dst,
                                                   size_t n) {
  size_t i = (size_t)blockIdx.x * blockDim.x + threadIdx.x;
  if (i < n) dst[i] = f2bf(src[i]);
}

// ---------------------------------------------------------------------------
// Transposing cast: WT[n][k] = (bf16) W[k][n]   (D_ x D_)
// ---------------------------------------------------------------------------
__global__ __launch_bounds__(256) void castT_kernel(const float* __restrict__ W,
                                                    __bf16* __restrict__ WT) {
  __shared__ float tile[32][33];
  const int tx = threadIdx.x & 31;
  const int ty = threadIdx.x >> 5;          // 0..7
  const int bx = blockIdx.x * 32;
  const int by = blockIdx.y * 32;
#pragma unroll
  for (int r = ty; r < 32; r += 8)
    tile[r][tx] = W[(size_t)(by + r) * D_ + bx + tx];
  __syncthreads();
#pragma unroll
  for (int r = ty; r < 32; r += 8)
    WT[(size_t)(bx + r) * D_ + by + tx] = f2bf(tile[tx][r]);
}

// ---------------------------------------------------------------------------
// GEMM: C[M_ x D_] (f32) = A[M_ x D_] (bf16, row-major) * BT[D_ x D_]^T
//   BT stored as BT[n][k] (row-major N x K) so B-fragments are contiguous.
// Workgroup tile 128x128, 8 waves of 32x64, K-step 32.
// Double-buffered LDS staged with GLOBAL_LOAD_ASYNC_TO_LDS_B128.
// ---------------------------------------------------------------------------
#define LDP 40   // padded LDS row stride (halfs): conflict-free b128 reads

__global__ __launch_bounds__(256) void gemm_bf16_kernel(
    const __bf16* __restrict__ A,
    const __bf16* __restrict__ BT,
    float* __restrict__ C) {
  __shared__ __bf16 As[2][128 * LDP];
  __shared__ __bf16 Bs[2][128 * LDP];

  const int tid  = threadIdx.x;
  const int wave = tid >> 5;
  const int lane = tid & 31;
  const int lr   = lane & 15;
  const int hi   = lane >> 4;

  const int bm = blockIdx.y * 128;
  const int bn = blockIdx.x * 128;

  const int wm = (wave >> 1) * 32;   // 0,32,64,96
  const int wn = (wave & 1) * 64;    // 0,64

  v8f acc[2][4];
#pragma unroll
  for (int i = 0; i < 2; i++)
#pragma unroll
    for (int j = 0; j < 4; j++)
      acc[i][j] = (v8f){0.f,0.f,0.f,0.f,0.f,0.f,0.f,0.f};

  // staging: 256 threads x 32B each per operand per K-step
  const int srow  = tid >> 1;
  const int stoff = srow * LDP + (tid & 1) * 16;
  const __bf16* Aptr = A  + (size_t)(bm + srow) * D_ + (tid & 1) * 16;
  const __bf16* Bptr = BT + (size_t)(bn + srow) * D_ + (tid & 1) * 16;

  int cur = 0;
  stage32(Aptr, As[0] + stoff);
  stage32(Bptr, Bs[0] + stoff);
  wait_async0();
  __syncthreads();

  for (int k0 = 0; k0 < D_; k0 += 32) {
    if (k0 + 32 < D_) {
      stage32(Aptr + k0 + 32, As[cur ^ 1] + stoff);
      stage32(Bptr + k0 + 32, Bs[cur ^ 1] + stoff);
      __builtin_prefetch(Aptr + k0 + 64, 0, 0);   // global_prefetch_b8
      __builtin_prefetch(Bptr + k0 + 64, 0, 0);
    }

    v16bf af[2], bfr[4];
#pragma unroll
    for (int mt = 0; mt < 2; mt++) {
      const __bf16* p = As[cur] + (wm + mt * 16 + lr) * LDP;
      af[mt] = load_frag(p + hi * 8, p + hi * 8 + 16);     // A layout
    }
#pragma unroll
    for (int nt = 0; nt < 4; nt++) {
      const __bf16* p = Bs[cur] + (wn + nt * 16 + lr) * LDP;
      bfr[nt] = load_frag(p + hi * 8, p + hi * 8 + 16);   // B layout
    }
#pragma unroll
    for (int mt = 0; mt < 2; mt++)
#pragma unroll
      for (int nt = 0; nt < 4; nt++)
        acc[mt][nt] = wmma_bf16(af[mt], bfr[nt], acc[mt][nt]);

    // Our frag ds_loads are drained by the WMMAs above (dscnt RAW), so after
    // this wait+barrier it is safe for everyone to overwrite buffer `cur`.
    wait_async0();
    __syncthreads();
    cur ^= 1;
  }

#pragma unroll
  for (int mt = 0; mt < 2; mt++)
#pragma unroll
    for (int nt = 0; nt < 4; nt++) {
      const int row0 = bm + wm + mt * 16 + hi * 8;
      const int col  = bn + wn + nt * 16 + lr;
      float* cp = C + (size_t)row0 * D_ + col;
#pragma unroll
      for (int r = 0; r < 8; r++) cp[(size_t)r * D_] = acc[mt][nt][r];
    }
}

// ---------------------------------------------------------------------------
// RoPE + layout permutes.  Input Qf/Kf/Vf are (B*L, D) row-major f32.
// Head-major layouts: (B,H,L,hd).  Vt is transposed per head: (B,H,hd,L).
// ---------------------------------------------------------------------------
__device__ __forceinline__ float rope_val(const float* src, int d, int l) {
  const int fi = d & 63;
  const float inv = __expf(-(float)(2 * fi) * (1.0f / (float)HD_) * 9.210340371976184f);
  const float ang = (float)l * inv;
  const float c = __cosf(ang), s = __sinf(ang);
  const float xv = src[d];
  const float xp = src[(d < 64) ? d + 64 : d - 64];
  return xv * c + ((d < 64) ? -xp : xp) * s;
}

__global__ __launch_bounds__(HD_) void rope_q_kernel(const float* __restrict__ Qf,
                                                     __bf16* __restrict__ Qb) {
  const int d = threadIdx.x, l = blockIdx.x, h = blockIdx.y, b = blockIdx.z;
  const float* src = Qf + ((size_t)(b * L_ + l)) * D_ + h * HD_;
  const float y = rope_val(src, d, l);
  Qb[((((size_t)b * H_ + h) * L_) + l) * HD_ + d] = f2bf(y);
}

__global__ __launch_bounds__(HD_) void rope_k_kernel(const float* __restrict__ Kf,
                                                     float* __restrict__ kout,
                                                     __bf16* __restrict__ Kb) {
  const int d = threadIdx.x, l = blockIdx.x, h = blockIdx.y, b = blockIdx.z;
  const float* src = Kf + ((size_t)(b * L_ + l)) * D_ + h * HD_;
  const float y = rope_val(src, d, l);
  const size_t hm = ((((size_t)b * H_ + h) * L_) + l) * HD_ + d;
  kout[hm] = y;
  Kb[hm]   = f2bf(y);
}

__global__ __launch_bounds__(HD_) void perm_v_kernel(const float* __restrict__ Vf,
                                                     float* __restrict__ vout,
                                                     __bf16* __restrict__ Vt) {
  const int d = threadIdx.x, l = blockIdx.x, h = blockIdx.y, b = blockIdx.z;
  const float v = Vf[((size_t)(b * L_ + l)) * D_ + h * HD_ + d];
  vout[((((size_t)b * H_ + h) * L_) + l) * HD_ + d] = v;
  Vt[((((size_t)b * H_ + h) * HD_) + d) * L_ + l] = f2bf(v);
}

// ---------------------------------------------------------------------------
// Flash attention: one wave per (b, h, 16-query tile). Online softmax.
//   Qb: (B,H,L,hd) bf16   Kb: (B,H,L,hd) bf16 (== BT layout for Q.K^T)
//   Vt: (B,H,hd,L) bf16 (== BT layout for P.V)
//   Of: (B*L, D) f32  (row-major, ready for the Wo GEMM)
// ---------------------------------------------------------------------------
__global__ __launch_bounds__(32) void flash_attn_kernel(
    const __bf16* __restrict__ Qb,
    const __bf16* __restrict__ Kb,
    const __bf16* __restrict__ Vt,
    float* __restrict__ Of) {
  __shared__ __bf16 Ps[16 * LDP];   // P tile, C-layout -> A-layout transpose

  const int lane = threadIdx.x;
  const int lr = lane & 15, hi = lane >> 4;
  const int qt = blockIdx.x, h = blockIdx.y, b = blockIdx.z;

  const size_t headOff = (((size_t)b * H_ + h) * L_) * HD_;
  const __bf16* Q  = Qb + headOff + (size_t)qt * 16 * HD_;
  const __bf16* Kh = Kb + headOff;
  const __bf16* Vh = Vt + (((size_t)b * H_ + h) * HD_) * L_;

  // Q fragments for hd=128: 4 K-chunks of 32, held in registers all along
  v16bf qf[4];
#pragma unroll
  for (int ks = 0; ks < 4; ks++) {
    const __bf16* p = Q + lr * HD_ + ks * 32;
    qf[ks] = load_frag(p + hi * 8, p + hi * 8 + 16);
  }

  v8f o[8];
#pragma unroll
  for (int t = 0; t < 8; t++) o[t] = (v8f){0.f,0.f,0.f,0.f,0.f,0.f,0.f,0.f};
  float m[8], lsum[8];
#pragma unroll
  for (int r = 0; r < 8; r++) { m[r] = -1e30f; lsum[r] = 0.f; }
  const float scale = 0.08838834764831845f;   // 1/sqrt(128)

  for (int kt = 0; kt < L_ / 32; kt++) {
    // ---- S = Q * K^T  (16 x 32 as two 16x16 C tiles) ----
    v8f s0 = (v8f){0.f,0.f,0.f,0.f,0.f,0.f,0.f,0.f};
    v8f s1 = (v8f){0.f,0.f,0.f,0.f,0.f,0.f,0.f,0.f};
#pragma unroll
    for (int ks = 0; ks < 4; ks++) {
      const __bf16* pk0 = Kh + (size_t)(kt * 32 + lr) * HD_ + ks * 32;
      const __bf16* pk1 = Kh + (size_t)(kt * 32 + 16 + lr) * HD_ + ks * 32;
      v16bf b0 = load_frag(pk0 + hi * 8, pk0 + hi * 8 + 16);
      v16bf b1 = load_frag(pk1 + hi * 8, pk1 + hi * 8 + 16);
      s0 = wmma_bf16(qf[ks], b0, s0);
      s1 = wmma_bf16(qf[ks], b1, s1);
    }

    // ---- online softmax (row = hi*8 + r, cols striped over 16 lanes) ----
    float alpha[8];
#pragma unroll
    for (int r = 0; r < 8; r++) {
      float v0 = s0[r] * scale, v1 = s1[r] * scale;
      float mx = fmaxf(v0, v1);
#pragma unroll
      for (int off = 1; off < 16; off <<= 1)
        mx = fmaxf(mx, __shfl_xor(mx, off, 32));
      const float newm = fmaxf(m[r], mx);
      const float p0 = __expf(v0 - newm);
      const float p1 = __expf(v1 - newm);
      alpha[r] = __expf(m[r] - newm);
      float rs = p0 + p1;
#pragma unroll
      for (int off = 1; off < 16; off <<= 1)
        rs += __shfl_xor(rs, off, 32);
      lsum[r] = lsum[r] * alpha[r] + rs;
      m[r] = newm;
      Ps[(hi * 8 + r) * LDP + lr]      = f2bf(p0);
      Ps[(hi * 8 + r) * LDP + 16 + lr] = f2bf(p1);
    }

#pragma unroll
    for (int t = 0; t < 8; t++)
#pragma unroll
      for (int r = 0; r < 8; r++) o[t][r] *= alpha[r];

    __syncthreads();
    const __bf16* pp = Ps + lr * LDP;
    v16bf pf = load_frag(pp + hi * 8, pp + hi * 8 + 16);   // A layout
    __syncthreads();

    // ---- O += P * V ----
#pragma unroll
    for (int nt = 0; nt < 8; nt++) {
      const __bf16* pv = Vh + (size_t)(nt * 16 + lr) * L_ + kt * 32;
      v16bf bv = load_frag(pv + hi * 8, pv + hi * 8 + 16);
      o[nt] = wmma_bf16(pf, bv, o[nt]);
    }
  }

  // ---- finalize: divide rows by softmax sum, store row-major (B*L, D) ----
  float inv[8];
#pragma unroll
  for (int r = 0; r < 8; r++) inv[r] = 1.0f / lsum[r];
#pragma unroll
  for (int nt = 0; nt < 8; nt++) {
    const size_t base =
        ((size_t)b * L_ + qt * 16 + hi * 8) * D_ + h * HD_ + nt * 16 + lr;
#pragma unroll
    for (int r = 0; r < 8; r++)
      Of[base + (size_t)r * D_] = o[nt][r] * inv[r];
  }
}

// ---------------------------------------------------------------------------
// Host launcher
// ---------------------------------------------------------------------------
extern "C" void kernel_launch(void* const* d_in, const int* in_sizes, int n_in,
                              void* d_out, int out_size, void* d_ws, size_t ws_size,
                              hipStream_t stream) {
  const float* x  = (const float*)d_in[0];
  const float* Wq = (const float*)d_in[1];
  const float* Wk = (const float*)d_in[2];
  const float* Wv = (const float*)d_in[3];
  const float* Wo = (const float*)d_in[4];

  float* out  = (float*)d_out;                 // (B, L, D)
  float* kout = out  + (size_t)M_ * D_;        // (B, H, L, hd)
  float* vout = kout + (size_t)M_ * D_;        // (B, H, L, hd)

  char* ws = (char*)d_ws;
  const size_t MB = 1024 * 1024;
  __bf16* xb  = (__bf16*)(ws +   0 * MB);   // 16 MB
  __bf16* WqT = (__bf16*)(ws +  16 * MB);   //  8 MB
  __bf16* WkT = (__bf16*)(ws +  24 * MB);   //  8 MB
  __bf16* WvT = (__bf16*)(ws +  32 * MB);   //  8 MB
  __bf16* WoT = (__bf16*)(ws +  40 * MB);   //  8 MB
  float*  Qf  = (float* )(ws +  48 * MB);   // 32 MB
  float*  Kf  = (float* )(ws +  80 * MB);   // 32 MB
  float*  Vf  = (float* )(ws + 112 * MB);   // 32 MB
  __bf16* Qb  = (__bf16*)(ws + 144 * MB);   // 16 MB
  __bf16* Kb  = (__bf16*)(ws + 160 * MB);   // 16 MB
  __bf16* Vt  = (__bf16*)(ws + 176 * MB);   // 16 MB
  float*  Of  = (float* )(ws + 192 * MB);   // 32 MB
  __bf16* Ob  = (__bf16*)(ws + 224 * MB);   // 16 MB  (total 240 MB)

  const size_t nX = (size_t)M_ * D_;

  cast_kernel<<<(unsigned)(nX / 256), 256, 0, stream>>>(x, xb, nX);
  dim3 tg(D_ / 32, D_ / 32);
  castT_kernel<<<tg, 256, 0, stream>>>(Wq, WqT);
  castT_kernel<<<tg, 256, 0, stream>>>(Wk, WkT);
  castT_kernel<<<tg, 256, 0, stream>>>(Wv, WvT);
  castT_kernel<<<tg, 256, 0, stream>>>(Wo, WoT);

  dim3 gg(D_ / 128, M_ / 128);
  gemm_bf16_kernel<<<gg, 256, 0, stream>>>(xb, WqT, Qf);
  gemm_bf16_kernel<<<gg, 256, 0, stream>>>(xb, WkT, Kf);
  gemm_bf16_kernel<<<gg, 256, 0, stream>>>(xb, WvT, Vf);

  dim3 rg(L_, H_, B_);
  rope_q_kernel<<<rg, HD_, 0, stream>>>(Qf, Qb);
  rope_k_kernel<<<rg, HD_, 0, stream>>>(Kf, kout, Kb);
  perm_v_kernel<<<rg, HD_, 0, stream>>>(Vf, vout, Vt);

  flash_attn_kernel<<<dim3(L_ / 16, H_, B_), 32, 0, stream>>>(Qb, Kb, Vt, Of);

  cast_kernel<<<(unsigned)(nX / 256), 256, 0, stream>>>(Of, Ob, nX);
  gemm_bf16_kernel<<<gg, 256, 0, stream>>>(Ob, WoT, out);
}
